// ATPSelfAttention_54142357733657
// MI455X (gfx1250) — matrix operand order repaired and checked by reading, not compiled, not run on hardware
//
#include <hip/hip_runtime.h>
#include <hip/hip_bf16.h>

// ---------------------------------------------------------------------------
// CDNA5 (gfx1250) bf16 WMMA self-attention pipeline with TDM tile staging
// ---------------------------------------------------------------------------

typedef __attribute__((ext_vector_type(16))) __bf16 v16bf;
typedef __attribute__((ext_vector_type(8)))  float  v8f;
typedef __attribute__((ext_vector_type(4)))  unsigned int v4u;
typedef __attribute__((ext_vector_type(4)))  int    v4i;
typedef __attribute__((ext_vector_type(8)))  int    v8i;

union BF16x16 { v16bf v; unsigned int u[8]; };
union F32x8   { v8f v; float f[8]; };

#define BB 2
#define SS 2048
#define DD 2048
#define HH 16
#define HD 128

#if defined(__AMDGCN__) && __has_builtin(__builtin_amdgcn_tensor_load_to_lds)
#define USE_TDM 1
#else
#define USE_TDM 0
#endif

__device__ __forceinline__ unsigned short f2bf(float f) {
    unsigned int u = __builtin_bit_cast(unsigned int, f);
    u += 0x7FFFu + ((u >> 16) & 1u);   // round-to-nearest-even
    return (unsigned short)(u >> 16);
}

__device__ __forceinline__ v8f wmma_bf16(const BF16x16& a, const BF16x16& b, v8f c) {
    return __builtin_amdgcn_wmma_f32_16x16x32_bf16(
        false, a.v, false, b.v, (short)0, c, false, false);
}

#if USE_TDM
// TDM 2-D tile load, bf16 elements. Loads tile (t0 x t1) from a row-major
// tensor (row stride = stride0 elements) into LDS at lds_off, inserting
// pad_amount dwords of LDS padding every pad_interval-coded dwords
// (pad codes per D# group1: interval 2<<code dwords, amount code+1 dwords).
__device__ __forceinline__ void tdm_load_2d_bf16(unsigned lds_off,
                                                 const unsigned short* gaddr,
                                                 unsigned t0, unsigned t1,
                                                 unsigned stride0,
                                                 unsigned pad_int_code,
                                                 unsigned pad_amt_code) {
    unsigned long long ga = (unsigned long long)(uintptr_t)gaddr;
    v4u g0;
    g0[0] = 1u;                                             // count=1, user mode
    g0[1] = lds_off;                                        // lds_addr (bytes)
    g0[2] = (unsigned)ga;                                   // global_addr[31:0]
    g0[3] = (unsigned)((ga >> 32) & 0x1FFFFFFu) | (2u << 30); // [56:32] | type=2
    v8i g1;
    unsigned w0 = (1u << 16)                                // data_size = 2 bytes
                | (1u << 20)                                // pad_enable
                | (pad_int_code << 22)
                | (pad_amt_code << 25);
    g1[0] = (int)w0;
    g1[1] = (int)((t0 & 0xFFFFu) << 16);                    // tensor_dim0[15:0]
    g1[2] = (int)(((t0 >> 16) & 0xFFFFu) | ((t1 & 0xFFFFu) << 16)); // td0 hi | td1 lo
    g1[3] = (int)(((t1 >> 16) & 0xFFFFu) | ((t0 & 0xFFFFu) << 16)); // td1 hi | tile_dim0
    g1[4] = (int)(t1 & 0xFFFFu);                            // tile_dim1 (tile_dim2=0)
    g1[5] = (int)stride0;                                   // tensor_dim0_stride lo32
    g1[6] = 0;
    g1[7] = 0;
    v4i z4 = (v4i)0;
#if defined(__clang_major__) && (__clang_major__ >= 23)
    v8i z8 = (v8i)0;
    __builtin_amdgcn_tensor_load_to_lds(g0, g1, z4, z4, z8, 0);
#else
    __builtin_amdgcn_tensor_load_to_lds(g0, g1, z4, z4, 0);
#endif
}

__device__ __forceinline__ void tdm_wait0() {
#if __has_builtin(__builtin_amdgcn_s_wait_tensorcnt)
    __builtin_amdgcn_s_wait_tensorcnt(0);
#else
    asm volatile("s_wait_tensorcnt 0x0" ::: "memory");
#endif
}
#endif  // USE_TDM

// ---------------------------------------------------------------------------
// fp32 -> bf16 conversion (vectorized x4)
// ---------------------------------------------------------------------------
__global__ void convert_f32_bf16(const float* __restrict__ in,
                                 unsigned short* __restrict__ out, int n4) {
    int i = blockIdx.x * blockDim.x + threadIdx.x;
    if (i < n4) {
        float4 f = ((const float4*)in)[i];
        unsigned int lo = (unsigned int)f2bf(f.x) | ((unsigned int)f2bf(f.y) << 16);
        unsigned int hi = (unsigned int)f2bf(f.z) | ((unsigned int)f2bf(f.w) << 16);
        ((uint2*)out)[i] = make_uint2(lo, hi);
    }
}

// ---------------------------------------------------------------------------
// Tiled bf16 GEMM. Block tile 128(M) x 64(N), K-step 32. 256 threads = 8 waves
// in a 4x2 wave grid; each wave owns a 32x32 tile = 2x2 WMMA accumulators.
// A tile staged by the Tensor Data Mover (row-major, padded rows); B tile
// transposed through VGPRs (overlaps the in-flight TDM transfer).
// MODE 0: C = x * w_qkv + b_qkv, scatter into q/k/v [B,H,S,HD] bf16
// MODE 1: C = ctx * w_dense + b_dense, write fp32 row-major [4096][2048]
// ---------------------------------------------------------------------------
template <int MODE>
__global__ __launch_bounds__(256)
void gemm_bf16_kernel(const unsigned int* __restrict__ A,   // bf16 pairs, [M][1024]
                      const unsigned int* __restrict__ Bw,  // bf16 pairs, [2048][Ntot/2]
                      const float* __restrict__ bias,
                      unsigned short* __restrict__ qo,
                      unsigned short* __restrict__ ko,
                      unsigned short* __restrict__ vo,
                      float* __restrict__ outp,
                      int Ntot) {
    __shared__ unsigned int   Au[128 * 20];      // 128 rows x 16 dwords (+4 pad)
    __shared__ unsigned short Bs[64 * 40];       // [n][k] transposed, 32 bf16 (+8 pad)

    const int tid  = threadIdx.x;
    const int w    = tid >> 5;
    const int lane = tid & 31;
    const int mw   = w & 3;
    const int nw   = w >> 2;
    const int lm   = lane & 15;
    const int lh   = lane >> 4;
    const int M0   = blockIdx.x * 128;
    const int N0   = blockIdx.y * 64;
    const int NtotU = Ntot >> 1;

    v8f acc[2][2];
    v8f zero = {0.f, 0.f, 0.f, 0.f, 0.f, 0.f, 0.f, 0.f};
#pragma unroll
    for (int i = 0; i < 2; ++i)
#pragma unroll
        for (int j = 0; j < 2; ++j) acc[i][j] = zero;

    for (int kk = 0; kk < DD; kk += 32) {
        __syncthreads();
#if USE_TDM
        // ---- A tile 128 x 32 bf16 via Tensor Data Mover ----
        if (w == 0) {
            tdm_load_2d_bf16((unsigned)(uintptr_t)&Au[0],
                             (const unsigned short*)A + (size_t)M0 * DD + kk,
                             /*t0=*/32, /*t1=*/128, /*stride0=*/DD,
                             /*pad 16 dwords interval*/3, /*4 dwords*/3);
        }
#else
#pragma unroll
        for (int i = 0; i < 8; ++i) {
            int di = i * 256 + tid;
            int row = di >> 4, c2 = di & 15;
            Au[row * 20 + c2] = A[(size_t)(M0 + row) * (DD / 2) + (kk >> 1) + c2];
        }
#endif
        // ---- B tile: 32(K) x 64(N) bf16, transposed into [n][k] ----
#pragma unroll
        for (int i = 0; i < 4; ++i) {
            int p = i * 256 + tid;
            int kr = p >> 5, nc2 = p & 31;
            unsigned int val = Bw[(size_t)(kk + kr) * NtotU + (N0 >> 1) + nc2];
            int n = nc2 * 2;
            Bs[n * 40 + kr]       = (unsigned short)(val & 0xFFFFu);
            Bs[(n + 1) * 40 + kr] = (unsigned short)(val >> 16);
        }
#if USE_TDM
        tdm_wait0();           // TENSORcnt==0 (no-op for non-issuing waves)
#endif
        __syncthreads();

        // ---- fragments per ISA 16-bit A(16x32) / B(32x16) layouts ----
        BF16x16 af[2], bfr[2];
#pragma unroll
        for (int tm = 0; tm < 2; ++tm) {
            int row = mw * 32 + tm * 16 + lm;
            int kb2 = lh * 4;                       // (lane>=16 ? K base 8 : 0)/2
#pragma unroll
            for (int j = 0; j < 4; ++j) af[tm].u[j]     = Au[row * 20 + kb2 + j];
#pragma unroll
            for (int j = 0; j < 4; ++j) af[tm].u[4 + j] = Au[row * 20 + 8 + kb2 + j];
        }
#pragma unroll
        for (int tn = 0; tn < 2; ++tn) {
            int n = nw * 32 + tn * 16 + lm;
            int kh = lh * 8;                        // lanes 16-31 hold K=8..15,24..31
            const unsigned int* bp = (const unsigned int*)(Bs + n * 40 + kh);
#pragma unroll
            for (int j = 0; j < 8; ++j) bfr[tn].u[j] = bp[j + (j & 4)];
        }
#pragma unroll
        for (int tm = 0; tm < 2; ++tm)
#pragma unroll
            for (int tn = 0; tn < 2; ++tn)
                acc[tm][tn] = wmma_bf16(af[tm], bfr[tn], acc[tm][tn]);
    }

    // ---- epilogue ----
#pragma unroll
    for (int tm = 0; tm < 2; ++tm) {
#pragma unroll
        for (int tn = 0; tn < 2; ++tn) {
            int n = N0 + nw * 32 + tn * 16 + lm;
            float bv = bias[n];
            F32x8 a; a.v = acc[tm][tn];
            if (MODE == 0) {
                int h = n / (3 * HD);
                int r = n % (3 * HD);
                int which = r >> 7;
                int d = r & 127;
                unsigned short* dst = (which == 0) ? qo : ((which == 1) ? ko : vo);
#pragma unroll
                for (int vv = 0; vv < 8; ++vv) {
                    int m = M0 + mw * 32 + tm * 16 + vv + lh * 8;
                    int b = m >> 11, s = m & 2047;
                    dst[(size_t)((b * HH + h) * SS + s) * HD + d] = f2bf(a.f[vv] + bv);
                }
            } else {
#pragma unroll
                for (int vv = 0; vv < 8; ++vv) {
                    int m = M0 + mw * 32 + tm * 16 + vv + lh * 8;
                    outp[(size_t)m * DD + n] = a.f[vv] + bv;
                }
            }
        }
    }
}

// ---------------------------------------------------------------------------
// Flash attention: block = 8 waves = 128 query rows of one (b,h).
// K tile staged by TDM (row-major + LDS pad), V tile transposed through VGPRs
// while the TDM transfer is in flight; scores = 8 WMMAs, online softmax in
// registers, P re-layout through wave-private LDS, ctx = 8 WMMAs.
// ---------------------------------------------------------------------------
__global__ __launch_bounds__(256)
void attn_kernel(const unsigned short* __restrict__ qm,
                 const unsigned short* __restrict__ km,
                 const unsigned short* __restrict__ vm,
                 const float* __restrict__ amask,     // [B,1,1,S]
                 unsigned short* __restrict__ ctx) {  // [B,S,D] bf16
    __shared__ unsigned int   Kl[32 * 68];    // K tile [key][d], 64 dwords +4 pad
    __shared__ unsigned short Vl[128 * 34];   // V tile transposed [d][key], +2 pad
    __shared__ unsigned int   Pl[8 * 256];    // per-wave P scratch, 16x32 bf16

    const int tid  = threadIdx.x;
    const int w    = tid >> 5;
    const int lane = tid & 31;
    const int lm   = lane & 15;
    const int lh   = lane >> 4;
    const int q0   = blockIdx.x * 128;
    const int h    = blockIdx.y;
    const int b    = blockIdx.z;
    const int qbase = q0 + w * 16;

    const size_t head = (size_t)(b * HH + h) * SS * HD;
    const unsigned int* qg = (const unsigned int*)(qm + head);   // stride 64 dwords
    const unsigned short* kg16 = km + head;
    const unsigned short* vg = vm + head;

    // Q fragments for this wave's 16 rows: 4 chunks of K=32 along HD=128
    BF16x16 aQ[4];
    {
        const unsigned int* qrow = qg + (size_t)(qbase + lm) * (HD / 2);
#pragma unroll
        for (int c = 0; c < 4; ++c) {
            int kb2 = c * 16 + lh * 4;
#pragma unroll
            for (int j = 0; j < 4; ++j) aQ[c].u[j]     = qrow[kb2 + j];
#pragma unroll
            for (int j = 0; j < 4; ++j) aQ[c].u[4 + j] = qrow[8 + kb2 + j];
        }
    }

    v8f cacc[8];
    float mrun[8], lrun[8], corrv[8];
    v8f zero = {0.f, 0.f, 0.f, 0.f, 0.f, 0.f, 0.f, 0.f};
#pragma unroll
    for (int i = 0; i < 8; ++i) { cacc[i] = zero; mrun[i] = -3.0e38f; lrun[i] = 0.f; }

    const float scale = 0.0883883476483184f;  // 1/sqrt(128)
    const int kend = q0 + 128;                // causal: keys <= max q row in block

    for (int kt = 0; kt < kend; kt += 32) {
        __syncthreads();
#if USE_TDM
        // ---- K tile 32 x 128 bf16 via Tensor Data Mover ----
        if (w == 0) {
            tdm_load_2d_bf16((unsigned)(uintptr_t)&Kl[0],
                             kg16 + (size_t)kt * HD,
                             /*t0=*/HD, /*t1=*/32, /*stride0=*/HD,
                             /*pad 64 dwords interval*/5, /*4 dwords*/3);
        }
#else
#pragma unroll
        for (int i = 0; i < 8; ++i) {
            int p = i * 256 + tid;
            int row = p >> 6, c2 = p & 63;
            Kl[row * 68 + c2] =
                ((const unsigned int*)kg16)[(size_t)(kt + row) * (HD / 2) + c2];
        }
#endif
        // ---- cooperative V tile load, transposed [d][key] ----
#pragma unroll
        for (int i = 0; i < 8; ++i) {
            int p = i * 256 + tid;
            int key = p >> 6, dp = (p & 63) * 2;
            unsigned int val =
                *(const unsigned int*)(vg + (size_t)(kt + key) * HD + dp);
            Vl[dp * 34 + key]       = (unsigned short)(val & 0xFFFFu);
            Vl[(dp + 1) * 34 + key] = (unsigned short)(val >> 16);
        }
#if USE_TDM
        tdm_wait0();
#endif
        __syncthreads();

        // ---- scores: S[16x32] = Q(16x128) . K^T, 2 n-tiles x 4 k-chunks ----
        v8f sacc[2] = {zero, zero};
#pragma unroll
        for (int nt = 0; nt < 2; ++nt) {
            int key = nt * 16 + lm;
#pragma unroll
            for (int c = 0; c < 4; ++c) {
                BF16x16 bK;
                int kb2 = c * 16 + lh * 4;   // d = c*32 + lh*8 + 2(j + (j&4))
#pragma unroll
                for (int j = 0; j < 8; ++j) bK.u[j] = Kl[key * 68 + kb2 + j + (j & 4)];
                sacc[nt] = wmma_bf16(aQ[c], bK, sacc[nt]);
            }
        }

        // ---- causal mask + attention_mask + online softmax ----
        float mcol0 = amask[(size_t)b * SS + kt + lm];
        float mcol1 = amask[(size_t)b * SS + kt + 16 + lm];
        F32x8 s0, s1; s0.v = sacc[0]; s1.v = sacc[1];
        unsigned short* Pu16 = (unsigned short*)&Pl[w * 256];
#pragma unroll
        for (int vv = 0; vv < 8; ++vv) {
            int lr   = vv + lh * 8;
            int qrow = qbase + lr;
            int key0 = kt + lm, key1 = kt + 16 + lm;
            float x0 = ((key0 <= qrow) ? s0.f[vv] * scale : -10000.0f) + mcol0;
            float x1 = ((key1 <= qrow) ? s1.f[vv] * scale : -10000.0f) + mcol1;
            float rm = fmaxf(x0, x1);
            rm = fmaxf(rm, __shfl_xor(rm, 1, 16));
            rm = fmaxf(rm, __shfl_xor(rm, 2, 16));
            rm = fmaxf(rm, __shfl_xor(rm, 4, 16));
            rm = fmaxf(rm, __shfl_xor(rm, 8, 16));
            float mnew = fmaxf(mrun[vv], rm);
            float corr = __expf(mrun[vv] - mnew);
            float p0 = __expf(x0 - mnew);
            float p1 = __expf(x1 - mnew);
            float rs = p0 + p1;
            rs += __shfl_xor(rs, 1, 16);
            rs += __shfl_xor(rs, 2, 16);
            rs += __shfl_xor(rs, 4, 16);
            rs += __shfl_xor(rs, 8, 16);
            lrun[vv] = lrun[vv] * corr + rs;
            mrun[vv] = mnew;
            corrv[vv] = corr;
            // P tile (bf16) to wave-private LDS in [row][key] order
            Pu16[lr * 32 + lm]      = f2bf(p0);
            Pu16[lr * 32 + 16 + lm] = f2bf(p1);
        }
        // rescale running context by correction factors (rows per C-layout)
#pragma unroll
        for (int dn = 0; dn < 8; ++dn) {
            F32x8 a; a.v = cacc[dn];
#pragma unroll
            for (int vv = 0; vv < 8; ++vv) a.f[vv] *= corrv[vv];
            cacc[dn] = a.v;
        }
        // wave-private LDS round trip: only need DS-counter drain, no barrier
        asm volatile("s_wait_dscnt 0" ::: "memory");

        // ---- P as A-fragment (16x32) ----
        BF16x16 pA;
        {
            const unsigned int* Pw = &Pl[w * 256];
            int kb2 = lh * 4;
#pragma unroll
            for (int j = 0; j < 4; ++j) pA.u[j]     = Pw[lm * 16 + kb2 + j];
#pragma unroll
            for (int j = 0; j < 4; ++j) pA.u[4 + j] = Pw[lm * 16 + 8 + kb2 + j];
        }
        // ---- ctx += P(16x32) . V(32x128), 8 n-tiles ----
#pragma unroll
        for (int dn = 0; dn < 8; ++dn) {
            BF16x16 bV;
            int d  = dn * 16 + lm;
            int kh = lh * 8;
            const unsigned short* vp = Vl + d * 34 + kh;
#pragma unroll
            for (int j = 0; j < 8; ++j)
                bV.u[j] = *(const unsigned int*)(vp + 2 * (j + (j & 4)));
            cacc[dn] = wmma_bf16(pA, bV, cacc[dn]);
        }
    }

    // ---- normalize and store ctx as [B,S,D] bf16 ----
    float rinv[8];
#pragma unroll
    for (int vv = 0; vv < 8; ++vv) rinv[vv] = 1.0f / lrun[vv];
#pragma unroll
    for (int dn = 0; dn < 8; ++dn) {
        F32x8 a; a.v = cacc[dn];
        int d = dn * 16 + lm;
#pragma unroll
        for (int vv = 0; vv < 8; ++vv) {
            int qrow = qbase + vv + lh * 8;
            ctx[(size_t)(b * SS + qrow) * DD + h * HD + d] = f2bf(a.f[vv] * rinv[vv]);
        }
    }
}

// ---------------------------------------------------------------------------
// Launch
// ---------------------------------------------------------------------------
extern "C" void kernel_launch(void* const* d_in, const int* in_sizes, int n_in,
                              void* d_out, int out_size, void* d_ws, size_t ws_size,
                              hipStream_t stream) {
    const float* x       = (const float*)d_in[0];
    const float* amask   = (const float*)d_in[1];
    const float* w_qkv   = (const float*)d_in[2];
    const float* b_qkv   = (const float*)d_in[3];
    const float* w_dense = (const float*)d_in[4];
    const float* b_dense = (const float*)d_in[5];
    float* out = (float*)d_out;

    char* ws = (char*)d_ws;
    unsigned short* xbf   = (unsigned short*)(ws + 0);               // 16 MiB
    unsigned short* wqkvb = (unsigned short*)(ws + 16777216);        // 24 MiB
    unsigned short* wdenb = (unsigned short*)(ws + 41943040);        // 8 MiB
    unsigned short* qb    = (unsigned short*)(ws + 50331648);        // 16 MiB
    unsigned short* kb    = (unsigned short*)(ws + 67108864);        // 16 MiB
    unsigned short* vb    = (unsigned short*)(ws + 83886080);        // 16 MiB
    unsigned short* ctxb  = (unsigned short*)(ws + 100663296);       // 16 MiB

    // 1) fp32 -> bf16 converts
    {
        int n4 = (BB * SS * DD) / 4;
        convert_f32_bf16<<<(n4 + 255) / 256, 256, 0, stream>>>(x, xbf, n4);
        n4 = (DD * 3 * DD) / 4;
        convert_f32_bf16<<<(n4 + 255) / 256, 256, 0, stream>>>(w_qkv, wqkvb, n4);
        n4 = (DD * DD) / 4;
        convert_f32_bf16<<<(n4 + 255) / 256, 256, 0, stream>>>(w_dense, wdenb, n4);
    }
    // 2) fused QKV projection -> q/k/v [B,H,S,HD] bf16
    {
        dim3 grid((BB * SS) / 128, (3 * DD) / 64);
        gemm_bf16_kernel<0><<<grid, 256, 0, stream>>>(
            (const unsigned int*)xbf, (const unsigned int*)wqkvb, b_qkv,
            qb, kb, vb, nullptr, 3 * DD);
    }
    // 3) flash attention -> ctx [B,S,D] bf16
    {
        dim3 grid(SS / 128, HH, BB);
        attn_kernel<<<grid, 256, 0, stream>>>(qb, kb, vb, amask, ctxb);
    }
    // 4) output projection -> out fp32 [B,S,D]
    {
        dim3 grid((BB * SS) / 128, DD / 64);
        gemm_bf16_kernel<1><<<grid, 256, 0, stream>>>(
            (const unsigned int*)ctxb, (const unsigned int*)wdenb, b_dense,
            nullptr, nullptr, nullptr, out, DD);
    }
}
